// CLIPAttention_11467562681048
// MI455X (gfx1250) — matrix operand order repaired and checked by reading, not compiled, not run on hardware
//
#include <hip/hip_runtime.h>

typedef __bf16 bf16;
typedef __attribute__((ext_vector_type(16))) __bf16 v16bf;
typedef __attribute__((ext_vector_type(8)))  float  v8f;
typedef __attribute__((ext_vector_type(4)))  unsigned v4u;
typedef __attribute__((ext_vector_type(8)))  int      v8i;
typedef __attribute__((ext_vector_type(4)))  int      v4i;

#define S_LEN  577
#define E_DIM  1024
#define B_SZ   16
#define H_CNT  16
#define D_HEAD 64
#define M_ROWS (B_SZ * S_LEN)   /* 9232 = 577*16, multiple of 16 */
#define SP     608              /* padded S for K dim (19*32)    */
#define P_ROWS 640              /* padded P rows for 64-row tiles */

#if defined(__has_builtin)
#if __has_builtin(__builtin_amdgcn_tensor_load_to_lds) && \
    __has_builtin(__builtin_amdgcn_s_wait_tensorcnt)
#define HAVE_TDM 1
#endif
#endif
#ifndef HAVE_TDM
#define HAVE_TDM 0
#endif

// --------------------------------------------------------------- fragments
// Row-major LDS tiles; fragment loads are two contiguous ds_load_b128 per lane.
// A (16x32 bf16): lane<16 -> M=lane, K 0..15; lane>=16 -> M=lane-16, K 16..31.
__device__ __forceinline__ v16bf frag_a(const bf16* base, int m, int half) {
  union { uint4 q[2]; v16bf v; } r;
  const bf16* p = base + m * 32 + half * 16;
  r.q[0] = *reinterpret_cast<const uint4*>(p);
  r.q[1] = *reinterpret_cast<const uint4*>(p + 8);
  return r.v;
}
// B (32x16 bf16): lane<16 -> N=lane, K 0..15; lane>=16 -> N=lane-16, K 16..31.
__device__ __forceinline__ v16bf frag_b(const bf16* base, int n, int half) {
  union { uint4 q[2]; v16bf v; } r;
  const bf16* p = base + n * 32 + half * 16;
  r.q[0] = *reinterpret_cast<const uint4*>(p);
  r.q[1] = *reinterpret_cast<const uint4*>(p + 8);
  return r.v;
}

// ---------------------------------------------------------------- utilities

__global__ __launch_bounds__(256) void f32_to_bf16_kernel(
    const float* __restrict__ x, bf16* __restrict__ y, size_t n) {
  size_t i = (size_t)blockIdx.x * blockDim.x + threadIdx.x;
  size_t stride = (size_t)gridDim.x * blockDim.x;
  for (; i < n; i += stride) y[i] = (bf16)x[i];
}

// One wave per row: P[h][i][:] = softmax(share_bias[h][i][:]) in bf16,
// zero-padded to SP cols / P_ROWS rows. The rank-1 score term is constant
// along the softmax axis, so it cancels exactly -> P is batch independent.
__global__ __launch_bounds__(256) void softmax_rows_kernel(
    const float* __restrict__ bias, bf16* __restrict__ P) {
  const int row  = blockIdx.x * 8 + (threadIdx.x >> 5);
  const int lane = threadIdx.x & 31;
  if (row >= H_CNT * P_ROWS) return;
  const int h = row / P_ROWS;
  const int i = row - h * P_ROWS;
  bf16* dst = P + ((size_t)h * P_ROWS + i) * SP;
  if (i >= S_LEN) {
    for (int j = lane; j < SP; j += 32) dst[j] = (bf16)0.0f;
    return;
  }
  const float* src = bias + ((size_t)h * S_LEN + i) * S_LEN;
  float mx = -3.4e38f;
  for (int j = lane; j < S_LEN; j += 32) mx = fmaxf(mx, src[j]);
#pragma unroll
  for (int o = 16; o > 0; o >>= 1) mx = fmaxf(mx, __shfl_xor(mx, o, 32));
  float sum = 0.0f;
  for (int j = lane; j < S_LEN; j += 32) sum += __expf(src[j] - mx);
#pragma unroll
  for (int o = 16; o > 0; o >>= 1) sum += __shfl_xor(sum, o, 32);
  const float inv = 1.0f / sum;
  for (int j = lane; j < S_LEN; j += 32) dst[j] = (bf16)(__expf(src[j] - mx) * inv);
  for (int j = S_LEN + lane; j < SP; j += 32) dst[j] = (bf16)0.0f;
}

// ------------------------------------------------- projection GEMM (x @ W.T + b)
// A: [M_ROWS, E_DIM] bf16 rm; W: [N=E_DIM, K=E_DIM] bf16 rm.
// Block 128x128, 8 waves, wave = 64x32 (4x2 WMMA tiles), K step 32.
// Double-buffered LDS, one barrier per K step; global loads staged in VGPRs
// and overlapped with the WMMA stream.
// MODE 0: out = Vt bf16 [B,H,D,SP] (transposed, s-contiguous). MODE 1: f32 [M,E].
template <int MODE>
__global__ __launch_bounds__(256) void gemm_rowwt_kernel(
    const bf16* __restrict__ A, const bf16* __restrict__ W,
    const float* __restrict__ bias, void* __restrict__ outp) {
  __shared__ __align__(16) bf16 ldsA[2][128 * 32];
  __shared__ __align__(16) bf16 ldsB[2][128 * 32];
  const int tid  = threadIdx.x;
  const int lane = tid & 31;
  const int wave = tid >> 5;
  const int wm   = wave & 1;
  const int wn   = wave >> 1;
  const int m0   = blockIdx.x * 128;
  const int n0   = blockIdx.y * 128;
  const int half = lane >> 4;
  const int lm   = lane & 15;

  v8f acc[4][2];
#pragma unroll
  for (int i = 0; i < 4; i++)
#pragma unroll
    for (int j = 0; j < 2; j++)
#pragma unroll
      for (int r = 0; r < 8; r++) acc[i][j][r] = 0.0f;

  uint4 sa[2], sb[2];
  auto stage_load = [&](int kb) {
#pragma unroll
    for (int rr = 0; rr < 2; rr++) {
      const int q = tid + rr * 256;  // 512 chunks of 8 bf16 per 128x32 tile
      const int mq = q >> 2, c = q & 3;
      uint4 av = make_uint4(0u, 0u, 0u, 0u);
      const int grow = m0 + mq;
      if (grow < M_ROWS)
        av = *reinterpret_cast<const uint4*>(A + (size_t)grow * E_DIM + kb + c * 8);
      sa[rr] = av;
      sb[rr] = *reinterpret_cast<const uint4*>(W + (size_t)(n0 + mq) * E_DIM + kb + c * 8);
    }
  };
  auto stage_store = [&](int buf) {
#pragma unroll
    for (int rr = 0; rr < 2; rr++) {
      const int q = tid + rr * 256;
      const int mq = q >> 2, c = q & 3;
      *reinterpret_cast<uint4*>(&ldsA[buf][mq * 32 + c * 8]) = sa[rr];
      *reinterpret_cast<uint4*>(&ldsB[buf][mq * 32 + c * 8]) = sb[rr];
    }
  };

  stage_load(0);
  stage_store(0);
  __syncthreads();
  int p = 0;
  for (int kb = 0; kb < E_DIM; kb += 32) {
    const bool more = (kb + 32) < E_DIM;
    if (more) stage_load(kb + 32);
    v16bf af[4], bfr[2];
#pragma unroll
    for (int i = 0; i < 4; i++)
      af[i] = frag_a(&ldsA[p][0], wm * 64 + i * 16 + lm, half);
#pragma unroll
    for (int j = 0; j < 2; j++)
      bfr[j] = frag_b(&ldsB[p][0], wn * 32 + j * 16 + lm, half);
#pragma unroll
    for (int i = 0; i < 4; i++)
#pragma unroll
      for (int j = 0; j < 2; j++)
        acc[i][j] = __builtin_amdgcn_wmma_f32_16x16x32_bf16(
            false, af[i], false, bfr[j], (short)0, acc[i][j], false, false);
    if (more) stage_store(p ^ 1);
    __syncthreads();
    p ^= 1;
  }

  const int ml = half * 8;  // C layout: elem r -> M = r + 8*(lane>>4), N = lane&15
#pragma unroll
  for (int i = 0; i < 4; i++) {
#pragma unroll
    for (int j = 0; j < 2; j++) {
      const int col = n0 + wn * 32 + j * 16 + lm;
      const float bv = bias[col];
#pragma unroll
      for (int r = 0; r < 8; r++) {
        const int row = m0 + wm * 64 + i * 16 + ml + r;
        if (row < M_ROWS) {
          const float v = acc[i][j][r] + bv;
          if (MODE == 0) {
            const int b = row / S_LEN;
            const int s = row - b * S_LEN;
            const int h = col >> 6, d = col & 63;
            ((bf16*)outp)[(((size_t)(b * H_CNT + h)) * D_HEAD + d) * SP + s] = (bf16)v;
          } else {
            ((float*)outp)[(size_t)row * E_DIM + col] = v;
          }
        }
      }
    }
  }
}

// ------------------------------------------- attention GEMM: ctx[b,h] = P[h] @ V[b,h]
// Block 64(M=s_i) x 64(N=d), 8 waves, wave = 16x32; K = SP over s_j.
// P tiles (64x32 bf16) are DMA'd into double-buffered LDS by the Tensor Data
// Mover (wave 0 issues tensor_load_to_lds, syncs with s_wait_tensorcnt); B
// fragments come straight from global (Vt is K-contiguous per lane).
__global__ __launch_bounds__(256) void gemm_attn_kernel(
    const bf16* __restrict__ P, const bf16* __restrict__ VT,
    bf16* __restrict__ ctx) {
  __shared__ __align__(16) bf16 ldsA[2][64 * 32];
  const int tid  = threadIdx.x;
  const int lane = tid & 31;
  const int wave = tid >> 5;
  const int wm   = wave & 3;   // m subtile (16 rows)
  const int wn   = wave >> 2;  // n half (32 cols)
  const int m0   = blockIdx.x * 64;
  const int bh   = blockIdx.y;  // b*16 + h
  const int b    = bh >> 4, h = bh & 15;
  const int half = lane >> 4;
  const int lm   = lane & 15;
  const bf16* Ph  = P + (size_t)h * P_ROWS * SP;
  const bf16* Vbh = VT + (size_t)bh * D_HEAD * SP;

  v8f acc[2];
#pragma unroll
  for (int j = 0; j < 2; j++)
#pragma unroll
    for (int r = 0; r < 8; r++) acc[j][r] = 0.0f;

  auto issue_tile = [&](int kb, int buf) {
#if HAVE_TDM
    if (wave == 0) {
      const unsigned lds_addr =
          (unsigned)(unsigned long long)(&ldsA[buf][0]);
      const unsigned long long ga =
          (unsigned long long)(Ph + (size_t)m0 * SP + kb);
      // D# group0: count=1, lds_addr, 57-bit global addr, type=2 (image).
      v4u g0;
      g0[0] = 1u;
      g0[1] = lds_addr;
      g0[2] = (unsigned)ga;
      g0[3] = ((unsigned)(ga >> 32) & 0x01FFFFFFu) | (2u << 30);
      // D# group1: mask=0, data_size=1(2B); tensor 608x640, tile 32x64,
      // dim0 stride 608. Field packing per ISA 8.4.
      v8i g1;
      g1[0] = (int)(1u << 16);                 // data_size=2B
      g1[1] = (int)((unsigned)SP << 16);       // tensor_dim0[15:0]
      g1[2] = (int)((unsigned)P_ROWS << 16);   // dim0[31:16]=0 | tensor_dim1[15:0]
      g1[3] = (int)(32u << 16);                // tensor_dim1[31:16]=0 | tile_dim0
      g1[4] = 64;                              // tile_dim1 | tile_dim2=0
      g1[5] = SP;                              // tensor_dim0_stride[31:0]
      g1[6] = 0;                               // stride hi | dim1_stride lo
      g1[7] = 0;
      v4i z4; z4[0] = 0; z4[1] = 0; z4[2] = 0; z4[3] = 0;
#if __clang_major__ >= 23
      v8i z8;
#pragma unroll
      for (int t = 0; t < 8; t++) z8[t] = 0;
      __builtin_amdgcn_tensor_load_to_lds(g0, g1, z4, z4, z8, 0);
#else
      __builtin_amdgcn_tensor_load_to_lds(g0, g1, z4, z4, 0);
#endif
    }
#else
    const int mq = tid >> 2, c = tid & 3;  // 256 chunks of 8 = 64x32 tile
    uint4 av = *reinterpret_cast<const uint4*>(Ph + (size_t)(m0 + mq) * SP + kb + c * 8);
    *reinterpret_cast<uint4*>(&ldsA[buf][mq * 32 + c * 8]) = av;
#endif
  };
  auto wait_tile = [&]() {
#if HAVE_TDM
    if (wave == 0) __builtin_amdgcn_s_wait_tensorcnt((short)0);
#endif
    __syncthreads();
  };

  issue_tile(0, 0);
  wait_tile();
  int p = 0;
  for (int kb = 0; kb < SP; kb += 32) {
    if (kb + 32 < SP) issue_tile(kb + 32, p ^ 1);
    const v16bf af = frag_a(&ldsA[p][0], wm * 16 + lm, half);
#pragma unroll
    for (int j = 0; j < 2; j++) {
      const int d  = wn * 32 + j * 16 + lm;
      const int kk = kb + half * 16;
      const v16bf bfr =
          *reinterpret_cast<const v16bf*>(Vbh + (size_t)d * SP + kk);
      acc[j] = __builtin_amdgcn_wmma_f32_16x16x32_bf16(
          false, af, false, bfr, (short)0, acc[j], false, false);
    }
    wait_tile();
    p ^= 1;
  }

  const int ml = half * 8;
#pragma unroll
  for (int j = 0; j < 2; j++) {
    const int col = h * D_HEAD + wn * 32 + j * 16 + lm;
#pragma unroll
    for (int r = 0; r < 8; r++) {
      const int row = m0 + wm * 16 + ml + r;
      if (row < S_LEN)
        ctx[(size_t)(b * S_LEN + row) * E_DIM + col] = (bf16)acc[j][r];
    }
  }
}

// ---------------------------------------------------------------- launcher

extern "C" void kernel_launch(void* const* d_in, const int* in_sizes, int n_in,
                              void* d_out, int out_size, void* d_ws, size_t ws_size,
                              hipStream_t stream) {
  const float* hs         = (const float*)d_in[0];
  const float* v_w        = (const float*)d_in[3];
  const float* v_b        = (const float*)d_in[4];
  const float* out_w      = (const float*)d_in[5];
  const float* out_b      = (const float*)d_in[6];
  const float* share_bias = (const float*)d_in[8];
  // d_in[1],[2] (q_w,q_b) and d_in[7] (share_key) cancel in the softmax.

  char* ws = (char*)d_ws;
  size_t off = 0;
  auto take = [&](size_t bytes) -> char* {
    char* p = ws + off;
    off = (off + bytes + 255) & ~(size_t)255;
    return p;
  };
  bf16* hsb = (bf16*)take((size_t)M_ROWS * E_DIM * 2);
  bf16* vwb = (bf16*)take((size_t)E_DIM * E_DIM * 2);
  bf16* owb = (bf16*)take((size_t)E_DIM * E_DIM * 2);
  bf16* P   = (bf16*)take((size_t)H_CNT * P_ROWS * SP * 2);
  bf16* VT  = (bf16*)take((size_t)B_SZ * H_CNT * D_HEAD * SP * 2);
  bf16* ctx = (bf16*)take((size_t)M_ROWS * E_DIM * 2);

  // Zero Vt so the s-padding [577,608) contributes 0 (P pad cols are 0 too).
  hipMemsetAsync(VT, 0, (size_t)B_SZ * H_CNT * D_HEAD * SP * 2, stream);

  const size_t nhs = (size_t)M_ROWS * E_DIM;
  const size_t nw  = (size_t)E_DIM * E_DIM;
  f32_to_bf16_kernel<<<dim3((unsigned)((nhs + 255) / 256)), 256, 0, stream>>>(hs, hsb, nhs);
  f32_to_bf16_kernel<<<dim3((unsigned)((nw + 255) / 256)), 256, 0, stream>>>(v_w, vwb, nw);
  f32_to_bf16_kernel<<<dim3((unsigned)((nw + 255) / 256)), 256, 0, stream>>>(out_w, owb, nw);

  softmax_rows_kernel<<<dim3(H_CNT * P_ROWS / 8), 256, 0, stream>>>(share_bias, P);

  gemm_rowwt_kernel<0><<<dim3((M_ROWS + 127) / 128, E_DIM / 128), 256, 0, stream>>>(
      hsb, vwb, v_b, (void*)VT);

  gemm_attn_kernel<<<dim3((S_LEN + 63) / 64, B_SZ * H_CNT), 256, 0, stream>>>(P, VT, ctx);

  gemm_rowwt_kernel<1><<<dim3((M_ROWS + 127) / 128, E_DIM / 128), 256, 0, stream>>>(
      ctx, owb, out_b, d_out);
}
